// MeshGNN_GAT_73220602462593
// MI455X (gfx1250) — matrix operand order repaired and checked by reading, not compiled, not run on hardware
//
#include <hip/hip_runtime.h>
#include <hip/hip_bf16.h>

#define N_NODES 100000
#define N_EDGES 1600000
#define IN_DIM 16
#define EDGE_DIM 4
#define HIDDEN 128
#define LAYERS 6
#define HEADS 8
#define CDIM 16
#define OUT_DIM 3
#define LN_EPS 1e-5f

typedef _Float16 half8 __attribute__((ext_vector_type(8)));
typedef _Float16 v16h  __attribute__((ext_vector_type(16)));
typedef float    v8f   __attribute__((ext_vector_type(8)));

// ---- order-preserving float <-> uint for atomicMax on floats ----
__device__ __forceinline__ unsigned fenc(float f) {
    unsigned u = __float_as_uint(f);
    return (u & 0x80000000u) ? ~u : (u | 0x80000000u);
}
__device__ __forceinline__ float fdec(unsigned e) {
    return (e & 0x80000000u) ? __uint_as_float(e & 0x7FFFFFFFu)
                             : __uint_as_float(~e);
}

__device__ __forceinline__ float wave_red_sum(float v) {
    for (int m = 16; m > 0; m >>= 1) v += __shfl_xor(v, m, 32);
    return v;
}

// Async global -> LDS copy, 16 bytes per active lane. Tracked by ASYNCcnt.
// The LDS-address operand is the 32-bit LDS byte offset: for generic pointers
// into __shared__, the low 32 bits ARE the LDS offset (flat LDS aperture).
__device__ __forceinline__ void async_copy_b128(void* lds_dst, const void* gsrc) {
    unsigned lds_addr = (unsigned)(unsigned long long)lds_dst;
    asm volatile("global_load_async_to_lds_b128 %0, %1, off"
                 :: "v"(lds_addr), "v"(gsrc)
                 : "memory");
}
__device__ __forceinline__ void wait_async_lds() {
    asm volatile("s_wait_asynccnt 0x0" ::: "memory");
}

// ---------------------------------------------------------------------------
// h0 = LayerNorm(relu(x @ W_in + b_in)); one wave per node, 4 channels/lane.
// ---------------------------------------------------------------------------
__global__ void k_input_proj(const float* __restrict__ x,
                             const float* __restrict__ Win,
                             const float* __restrict__ bin,
                             const float* __restrict__ g,
                             const float* __restrict__ b,
                             float* __restrict__ h) {
    int wave = (blockIdx.x * blockDim.x + threadIdx.x) >> 5;
    int lane = threadIdx.x & 31;
    if (wave >= N_NODES) return;
    const float* xr = x + (size_t)wave * IN_DIM;
    float y[4], s1 = 0.f, s2 = 0.f;
    for (int j = 0; j < 4; j++) {
        int c = lane * 4 + j;
        float acc = bin[c];
        for (int k = 0; k < IN_DIM; k++) acc += xr[k] * Win[k * HIDDEN + c];
        acc = acc > 0.f ? acc : 0.f;
        y[j] = acc; s1 += acc; s2 += acc * acc;
    }
    s1 = wave_red_sum(s1); s2 = wave_red_sum(s2);
    float mu = s1 * (1.f / HIDDEN);
    float var = s2 * (1.f / HIDDEN) - mu * mu;
    float rs = rsqrtf(var + LN_EPS);
    for (int j = 0; j < 4; j++) {
        int c = lane * 4 + j;
        h[(size_t)wave * HIDDEN + c] = (y[j] - mu) * rs * g[c] + b[c];
    }
}

// ---------------------------------------------------------------------------
// Out[nrows,128] = (relu?)(A[nrows,128] @ W[128,128] + bias), WMMA f16->f32.
// 256 threads = 8 wave32; 128 rows per block.
//  - A tile staged f32 via GLOBAL_LOAD_ASYNC_TO_LDS_B128 (ASYNCcnt path);
//    f32->f16 conversion happens once per wave while building fragments.
//  - W staged transposed [n][k] as f16 for contiguous B-fragment reads.
//  - All 4 K-step B fragments are loaded before the WMMA chain so the 8
//    ds_load_b128 overlap the matrix pipe (partial s_wait_dscnt).
// Note: async staging may over-read past the final partial block; callers
// guarantee A has >=48KB of valid workspace after it, and out-of-range rows
// are never stored (WMMA row m depends only on A row m).
// ---------------------------------------------------------------------------
template <bool BIAS_RELU>
__global__ void __launch_bounds__(256)
k_gemm128(const float* __restrict__ A, const float* __restrict__ W,
          const float* __restrict__ bias, float* __restrict__ Out,
          int nrows) {
    __shared__ __align__(16) float    fA[128 * 128];   // 64KB, [row][k] f32
    __shared__ __align__(16) _Float16 sW[128 * 128];   // 32KB, transposed [n][k]
    const int tid = threadIdx.x;
    const int blockRow = blockIdx.x * 128;

    // Async-stage the 128x512B activation tile: 4096 x 16B chunks, 16/thread.
    {
        const char* gbase = (const char*)(A + (size_t)blockRow * 128);
        char* lbase = (char*)fA;
        for (int c = tid; c < 4096; c += 256)
            async_copy_b128(lbase + c * 16, gbase + (size_t)c * 16);
    }
    // Stage W transposed as f16 (reused from L2 across all 782 blocks).
    for (int idx = tid; idx < 128 * 128; idx += 256) {
        int k = idx >> 7, n = idx & 127;
        sW[n * 128 + k] = (_Float16)W[idx];            // coalesced global read
    }
    wait_async_lds();
    __syncthreads();

    const int wv = tid >> 5, lane = tid & 31;
    const int m0 = wv * 16;
    const int lrow = lane & 15;     // row (A) / col (B,C)
    const int lhi  = lane >> 4;     // lane half selector

    // Build all 4 K-step A fragments once (f32 LDS -> f16 regs).
    v16h afrag[4];
    for (int kt = 0; kt < 4; kt++) {
        const int k0 = kt * 32;
        const float* ar = &fA[(m0 + lrow) * 128];
        const float4 a0 = *(const float4*)&ar[k0 + lhi * 8];
        const float4 a1 = *(const float4*)&ar[k0 + lhi * 8 + 4];
        const float4 a2 = *(const float4*)&ar[k0 + 16 + lhi * 8];
        const float4 a3 = *(const float4*)&ar[k0 + 16 + lhi * 8 + 4];
        afrag[kt][0]  = (_Float16)a0.x; afrag[kt][1]  = (_Float16)a0.y;
        afrag[kt][2]  = (_Float16)a0.z; afrag[kt][3]  = (_Float16)a0.w;
        afrag[kt][4]  = (_Float16)a1.x; afrag[kt][5]  = (_Float16)a1.y;
        afrag[kt][6]  = (_Float16)a1.z; afrag[kt][7]  = (_Float16)a1.w;
        afrag[kt][8]  = (_Float16)a2.x; afrag[kt][9]  = (_Float16)a2.y;
        afrag[kt][10] = (_Float16)a2.z; afrag[kt][11] = (_Float16)a2.w;
        afrag[kt][12] = (_Float16)a3.x; afrag[kt][13] = (_Float16)a3.y;
        afrag[kt][14] = (_Float16)a3.z; afrag[kt][15] = (_Float16)a3.w;
    }

    const bool fullBlock = (blockRow + 128) <= nrows;

    for (int nt = 0; nt < 8; nt++) {
        const int n0 = nt * 16;
        // Load all B fragments first: 8 ds_load_b128 in flight before WMMAs.
        const _Float16* wrow = &sW[(n0 + lrow) * 128 + lhi * 8];
        v16h bf[4];
        for (int kt = 0; kt < 4; kt++) {
            const half8 blo = *(const half8*)&wrow[kt * 32];
            const half8 bhi = *(const half8*)&wrow[kt * 32 + 16];
            for (int i = 0; i < 8; i++) { bf[kt][i] = blo[i]; bf[kt][8 + i] = bhi[i]; }
        }
        v8f acc = {0.f, 0.f, 0.f, 0.f, 0.f, 0.f, 0.f, 0.f};
        for (int kt = 0; kt < 4; kt++)
            acc = __builtin_amdgcn_wmma_f32_16x16x32_f16(
                false, afrag[kt], false, bf[kt], (short)0, acc, false, false);

        const int col = n0 + lrow;
        float bv = 0.0f;
        if (BIAS_RELU) bv = bias[col];
        if (fullBlock) {
            for (int i = 0; i < 8; i++) {
                int gr = blockRow + m0 + i + lhi * 8;  // C/D: M = i + 8*(lane>=16)
                float v = acc[i];
                if (BIAS_RELU) { v += bv; v = v > 0.f ? v : 0.f; }
                Out[(size_t)gr * 128 + col] = v;
            }
        } else {
            for (int i = 0; i < 8; i++) {
                int gr = blockRow + m0 + i + lhi * 8;
                if (gr < nrows) {
                    float v = acc[i];
                    if (BIAS_RELU) { v += bv; v = v > 0.f ? v : 0.f; }
                    Out[(size_t)gr * 128 + col] = v;
                }
            }
        }
    }
}

// ---------------------------------------------------------------------------
// Per (node, head): alpha_src/dst dots; reset m/denom; zero accumulator slice.
// ---------------------------------------------------------------------------
__global__ void k_node_alpha(const float* __restrict__ hp,
                             const float* __restrict__ a_s,
                             const float* __restrict__ a_d,
                             float* __restrict__ alpha_src,
                             float* __restrict__ alpha_dst,
                             unsigned* __restrict__ m_u,
                             float* __restrict__ denom,
                             float* __restrict__ out_acc) {
    int i = blockIdx.x * blockDim.x + threadIdx.x;
    if (i >= N_NODES * HEADS) return;
    int n = i >> 3, hh = i & 7;
    const float* row = hp + (size_t)n * HIDDEN + hh * CDIM;
    float ss = 0.f, sd = 0.f;
    for (int c = 0; c < CDIM; c++) {
        float v = row[c];
        ss += v * a_s[hh * CDIM + c];
        sd += v * a_d[hh * CDIM + c];
    }
    alpha_src[i] = ss; alpha_dst[i] = sd;
    m_u[i] = 0u;          // encoded "-inf" floor for atomicMax
    denom[i] = 0.f;
    float* oa = out_acc + (size_t)n * HIDDEN + hh * CDIM;
    for (int c = 0; c < CDIM; c++) oa[c] = 0.f;
}

// Fold We against att_edge: wea[d*8+h] = sum_c We[d, h*16+c] * a_e[h,c]
__global__ void k_wea(const float* __restrict__ We,
                      const float* __restrict__ ae,
                      float* __restrict__ wea) {
    int t = threadIdx.x;
    if (t >= EDGE_DIM * HEADS) return;
    int d = t >> 3, hh = t & 7;
    float s = 0.f;
    for (int c = 0; c < CDIM; c++)
        s += We[d * HIDDEN + hh * CDIM + c] * ae[hh * CDIM + c];
    wea[t] = s;
}

// Per (edge, head): leaky-relu logit; atomic segment max into m_u[dst].
__global__ void k_edge_logits(const int* __restrict__ ei,
                              const float* __restrict__ ea,
                              const float* __restrict__ wea,
                              const float* __restrict__ alpha_src,
                              const float* __restrict__ alpha_dst,
                              float* __restrict__ alpha,
                              unsigned* __restrict__ m_u) {
    long long i = (long long)blockIdx.x * blockDim.x + threadIdx.x;
    if (i >= (long long)N_EDGES * HEADS) return;
    int e = (int)(i >> 3), hh = (int)(i & 7);
    int s = ei[e], d = ei[N_EDGES + e];
    const float4 eav = *(const float4*)&ea[e * EDGE_DIM];
    float al = alpha_src[s * 8 + hh] + alpha_dst[d * 8 + hh];
    al += eav.x * wea[0 * 8 + hh] + eav.y * wea[1 * 8 + hh]
        + eav.z * wea[2 * 8 + hh] + eav.w * wea[3 * 8 + hh];
    al = al > 0.f ? al : 0.2f * al;            // leaky_relu slope 0.2
    alpha[i] = al;
    atomicMax(&m_u[d * 8 + hh], fenc(al));
}

__global__ void k_fix_m(const unsigned* __restrict__ m_u,
                        float* __restrict__ mf) {
    int i = blockIdx.x * blockDim.x + threadIdx.x;
    if (i >= N_NODES * HEADS) return;
    unsigned u = m_u[i];
    mf[i] = (u == 0u) ? 0.0f : fdec(u);        // no-edge segments -> 0
}

// Per (edge, head): a = exp(alpha - m[dst]) in place; atomic denom sum.
__global__ void k_edge_exp(const int* __restrict__ ei,
                           const float* __restrict__ mf,
                           float* __restrict__ alpha,
                           float* __restrict__ denom) {
    long long i = (long long)blockIdx.x * blockDim.x + threadIdx.x;
    if (i >= (long long)N_EDGES * HEADS) return;
    int e = (int)(i >> 3), hh = (int)(i & 7);
    int d = ei[N_EDGES + e];
    float a = __expf(alpha[i] - mf[d * 8 + hh]);
    alpha[i] = a;
    atomicAdd(&denom[d * 8 + hh], a);
}

// One wave per 4 edges: gather hp[src] (float4/lane), scale by softmax weight,
// atomic scatter into accumulator at dst. Node state (hp + acc + denom) fits
// in the 192MB L2, so these random accesses stay on-chip. Prefetch the next
// edge's source row to overlap gather latency (global_prefetch_b8).
#define EDGES_PER_WAVE 4
__global__ void k_scatter(const int* __restrict__ ei,
                          const float* __restrict__ hp,
                          const float* __restrict__ alpha,
                          const float* __restrict__ denom,
                          float* __restrict__ out_acc) {
    int wave = (blockIdx.x * blockDim.x + threadIdx.x) >> 5;
    int lane = threadIdx.x & 31;
    int ebase = wave * EDGES_PER_WAVE;
    if (ebase >= N_EDGES) return;
    int hh = lane >> 2;                         // 4 lanes per head
    for (int j = 0; j < EDGES_PER_WAVE; j++) {
        int e = ebase + j;
        if (e >= N_EDGES) return;
        if (j + 1 < EDGES_PER_WAVE && e + 1 < N_EDGES) {
            int sn = ei[e + 1];
            __builtin_prefetch(&hp[(size_t)sn * HIDDEN + lane * 4], 0, 1);
        }
        int s = ei[e], d = ei[N_EDGES + e];
        float w = alpha[(size_t)e * 8 + hh] / (denom[d * 8 + hh] + 1e-16f);
        const float4 v = *(const float4*)&hp[(size_t)s * HIDDEN + lane * 4];
        float* o = out_acc + (size_t)d * HIDDEN + lane * 4;
        atomicAdd(&o[0], v.x * w);
        atomicAdd(&o[1], v.y * w);
        atomicAdd(&o[2], v.z * w);
        atomicAdd(&o[3], v.w * w);
    }
}

// h_next = LayerNorm(h + acc + bg); in-place on acc buffer; wave per node.
__global__ void k_res_ln(const float* __restrict__ h,
                         const float* __restrict__ bg,
                         const float* __restrict__ g,
                         const float* __restrict__ b,
                         float* __restrict__ acc_out) {
    int wave = (blockIdx.x * blockDim.x + threadIdx.x) >> 5;
    int lane = threadIdx.x & 31;
    if (wave >= N_NODES) return;
    float y[4], s1 = 0.f, s2 = 0.f;
    for (int j = 0; j < 4; j++) {
        int c = lane * 4 + j;
        float v = h[(size_t)wave * HIDDEN + c] + acc_out[(size_t)wave * HIDDEN + c] + bg[c];
        y[j] = v; s1 += v; s2 += v * v;
    }
    s1 = wave_red_sum(s1); s2 = wave_red_sum(s2);
    float mu = s1 * (1.f / HIDDEN);
    float var = s2 * (1.f / HIDDEN) - mu * mu;
    float rs = rsqrtf(var + LN_EPS);
    for (int j = 0; j < 4; j++) {
        int c = lane * 4 + j;
        acc_out[(size_t)wave * HIDDEN + c] = (y[j] - mu) * rs * g[c] + b[c];
    }
}

// out = relu(t @ W2 + b2), W2 is 128x3 -> one thread per node.
__global__ void k_head2(const float* __restrict__ t,
                        const float* __restrict__ W2,
                        const float* __restrict__ b2,
                        float* __restrict__ out) {
    int n = blockIdx.x * blockDim.x + threadIdx.x;
    if (n >= N_NODES) return;
    float a0 = b2[0], a1 = b2[1], a2 = b2[2];
    const float* tr = t + (size_t)n * HIDDEN;
    for (int k = 0; k < HIDDEN; k++) {
        float v = tr[k];
        a0 += v * W2[k * 3 + 0];
        a1 += v * W2[k * 3 + 1];
        a2 += v * W2[k * 3 + 2];
    }
    out[n * 3 + 0] = a0 > 0.f ? a0 : 0.f;
    out[n * 3 + 1] = a1 > 0.f ? a1 : 0.f;
    out[n * 3 + 2] = a2 > 0.f ? a2 : 0.f;
}

extern "C" void kernel_launch(void* const* d_in, const int* in_sizes, int n_in,
                              void* d_out, int out_size, void* d_ws, size_t ws_size,
                              hipStream_t stream) {
    const float* x    = (const float*)d_in[0];
    const int*   ei   = (const int*)  d_in[1];
    const float* ea   = (const float*)d_in[2];
    const float* Win  = (const float*)d_in[3];
    const float* bin  = (const float*)d_in[4];
    const float* lnig = (const float*)d_in[5];
    const float* lnib = (const float*)d_in[6];
    const float* Wg   = (const float*)d_in[7];
    const float* a_s  = (const float*)d_in[8];
    const float* a_d  = (const float*)d_in[9];
    const float* We   = (const float*)d_in[10];
    const float* a_e  = (const float*)d_in[11];
    const float* bg   = (const float*)d_in[12];
    const float* lg   = (const float*)d_in[13];
    const float* lb   = (const float*)d_in[14];
    const float* W1   = (const float*)d_in[15];
    const float* b1   = (const float*)d_in[16];
    const float* W2   = (const float*)d_in[17];
    const float* b2   = (const float*)d_in[18];

    char* ws = (char*)d_ws;
    size_t off = 0;
    auto carve = [&](size_t bytes) -> void* {
        void* p = ws + off;
        off += (bytes + 255) & ~(size_t)255;
        return p;
    };
    float*    bufA  = (float*)   carve((size_t)N_NODES * HIDDEN * 4);
    float*    bufB  = (float*)   carve((size_t)N_NODES * HIDDEN * 4);
    float*    hp    = (float*)   carve((size_t)N_NODES * HIDDEN * 4);
    float*    alpha = (float*)   carve((size_t)N_EDGES * HEADS * 4);
    float*    asrc  = (float*)   carve((size_t)N_NODES * HEADS * 4);
    float*    adst  = (float*)   carve((size_t)N_NODES * HEADS * 4);
    unsigned* m_u   = (unsigned*)carve((size_t)N_NODES * HEADS * 4);
    float*    mf    = (float*)   carve((size_t)N_NODES * HEADS * 4);
    float*    den   = (float*)   carve((size_t)N_NODES * HEADS * 4);
    float*    wea   = (float*)   carve(EDGE_DIM * HEADS * 4);

    const dim3 blk(256);
    const int nodeWaveBlocks = (N_NODES * 32 + 255) / 256;
    const int nhBlocks       = (N_NODES * HEADS + 255) / 256;
    const int ehBlocks       = (int)(((long long)N_EDGES * HEADS + 255) / 256);
    const int scatterBlocks  = (N_EDGES + 8 * EDGES_PER_WAVE - 1) / (8 * EDGES_PER_WAVE);
    const int gemmBlocks     = (N_NODES + 127) / 128;

    k_input_proj<<<nodeWaveBlocks, blk, 0, stream>>>(x, Win, bin, lnig, lnib, bufA);

    float* h  = bufA;
    float* hn = bufB;
    for (int l = 0; l < LAYERS; l++) {
        k_gemm128<false><<<gemmBlocks, blk, 0, stream>>>(
            h, Wg + (size_t)l * HIDDEN * HIDDEN, nullptr, hp, N_NODES);
        k_node_alpha<<<nhBlocks, blk, 0, stream>>>(
            hp, a_s + l * HEADS * CDIM, a_d + l * HEADS * CDIM,
            asrc, adst, m_u, den, hn);
        k_wea<<<1, 32, 0, stream>>>(
            We + (size_t)l * EDGE_DIM * HIDDEN, a_e + l * HEADS * CDIM, wea);
        k_edge_logits<<<ehBlocks, blk, 0, stream>>>(ei, ea, wea, asrc, adst, alpha, m_u);
        k_fix_m<<<nhBlocks, blk, 0, stream>>>(m_u, mf);
        k_edge_exp<<<ehBlocks, blk, 0, stream>>>(ei, mf, alpha, den);
        k_scatter<<<scatterBlocks, blk, 0, stream>>>(ei, hp, alpha, den, hn);
        k_res_ln<<<nodeWaveBlocks, blk, 0, stream>>>(
            h, bg + l * HIDDEN, lg + l * HIDDEN, lb + l * HIDDEN, hn);
        float* tmp = h; h = hn; hn = tmp;
    }

    k_gemm128<true><<<gemmBlocks, blk, 0, stream>>>(h, W1, b1, hp, N_NODES);
    k_head2<<<(N_NODES + 255) / 256, blk, 0, stream>>>(hp, W2, b2, (float*)d_out);
}
